// GaussianLayer_63909113365137
// MI455X (gfx1250) — matrix operand order repaired and checked by reading, not compiled, not run on hardware
//
#include <hip/hip_runtime.h>
#include <hip/hip_bf16.h>

typedef _Float16 half_t;
typedef __attribute__((ext_vector_type(16))) _Float16 v16h;
typedef __attribute__((ext_vector_type(8)))  _Float16 v8h;
typedef __attribute__((ext_vector_type(8)))  float    v8f;

union Frag16 { v16h v; v8h h[2]; };

#define BATCH 4096
#define IN_F  4096
#define OUT_F 4096
#define NCOMP 512

// Block tile 128x128, K-step 64, padded LDS row stride 72 halfs (144 B).
#define KSTEP   64
#define LDS_STR 72

// ---------------------------------------------------------------------------
// Tensor Data Mover support (guarded: falls back to manual staging)
// ---------------------------------------------------------------------------
#if defined(__has_builtin)
#  if __has_builtin(__builtin_amdgcn_tensor_load_to_lds)
#    define USE_TDM 1
#  endif
#endif
#ifndef USE_TDM
#  define USE_TDM 0
#endif

__device__ inline void wait_tensor() {
#if defined(__has_builtin) && __has_builtin(__builtin_amdgcn_s_wait_tensorcnt)
    __builtin_amdgcn_s_wait_tensorcnt(0);
#else
    asm volatile("s_wait_tensorcnt 0x0" ::: "memory");
#endif
}

#if USE_TDM
typedef __attribute__((ext_vector_type(4))) unsigned int u32x4_t;
typedef __attribute__((ext_vector_type(8))) int          i32x8_t;
typedef __attribute__((ext_vector_type(4))) int          i32x4_t;

__device__ inline unsigned lds_u32(const void* p) {
    // addrspacecast(AS3 -> generic) keeps the LDS byte offset in the low 32 bits
    return (unsigned)(unsigned long long)(size_t)p;
}

// 2-D f16 tile load: TILE_K contiguous halfs per row, TILE_N rows, row stride
// `stride_elems` halfs in memory; LDS destination padded +4 DWORDs every
// 32 DWORDs (row of 64 halfs -> LDS stride 72 halfs). D# per ISA ch.8.
template <int TILE_K, int TILE_N, int PAD_IVL_CODE, int PAD_AMT_CODE>
__device__ inline void tdm_tile_f16(const half_t* g, unsigned lds_off, int stride_elems) {
    unsigned long long ga = (unsigned long long)(size_t)g;
    u32x4_t g0;
    g0[0] = 1u;                                              // count=1, user desc
    g0[1] = lds_off;                                         // lds_addr (bytes)
    g0[2] = (unsigned)(ga & 0xffffffffull);                  // global_addr[31:0]
    g0[3] = (unsigned)((ga >> 32) & 0x1ffffffull) | (2u << 30); // ga[56:32], type=2
    i32x8_t g1;
    g1[0] = (int)((1u << 16)                                 // data_size = 2 bytes
                | (1u << 20)                                 // pad_enable
                | ((unsigned)PAD_IVL_CODE << 22)             // pad every 2<<code DWORDs
                | ((unsigned)PAD_AMT_CODE << 25));           // pad (code+1) DWORDs
    g1[1] = (int)(((unsigned)TILE_K & 0xffffu) << 16);       // tensor_dim0[15:0]
    g1[2] = (int)((((unsigned)TILE_K >> 16) & 0xffffu)
                | (((unsigned)TILE_N & 0xffffu) << 16));     // dim0 hi | dim1 lo
    g1[3] = (int)((((unsigned)TILE_N >> 16) & 0xffffu)
                | ((unsigned)TILE_K << 16));                 // dim1 hi | tile_dim0
    g1[4] = (int)((unsigned)TILE_N);                         // tile_dim1 (tile_dim2=0)
    g1[5] = (int)(unsigned)stride_elems;                     // tensor_dim0_stride lo32
    g1[6] = 0;
    g1[7] = 0;
    i32x4_t z4 = {0, 0, 0, 0};                               // groups 2/3: NULL (2-D)
    i32x8_t z8 = {0, 0, 0, 0, 0, 0, 0, 0};
    // 6-arg form (this toolchain): (g0, g1, g2, g3, extra, cpol)
    __builtin_amdgcn_tensor_load_to_lds(g0, g1, z4, z4, z8, 0);
}
#endif // USE_TDM

// ---------------------------------------------------------------------------
// Manual staging helpers (global -> regs -> LDS), used for the f32 A matrix
// and as the no-TDM fallback for f16 tiles.
// ---------------------------------------------------------------------------
template <int Ktot>
__device__ inline void loadA_f32(const float* __restrict__ A, int m0, int k0, int t,
                                 float4 r[8]) {
#pragma unroll
    for (int q = 0; q < 8; ++q) {
        int f = t + 256 * q;               // 2048 float4 chunks: 128 rows x 16
        int row = f >> 4, c4 = f & 15;
        r[q] = *(const float4*)(A + (size_t)(m0 + row) * Ktot + k0 + c4 * 4);
    }
}
__device__ inline void storeA_f32(half_t* __restrict__ lds, int t, const float4 r[8]) {
#pragma unroll
    for (int q = 0; q < 8; ++q) {
        int f = t + 256 * q;
        int row = f >> 4, c4 = f & 15;
        half_t* d = &lds[row * LDS_STR + c4 * 4];
        d[0] = (half_t)r[q].x; d[1] = (half_t)r[q].y;
        d[2] = (half_t)r[q].z; d[3] = (half_t)r[q].w;
    }
}
template <int Ktot>
__device__ inline void load_f16_tile(const half_t* __restrict__ S, int r0, int k0, int t,
                                     v8h r[4]) {
#pragma unroll
    for (int q = 0; q < 4; ++q) {
        int c = t + 256 * q;               // 1024 v8h chunks: 128 rows x 8
        int row = c >> 3, c8 = c & 7;
        r[q] = *(const v8h*)(S + (size_t)(r0 + row) * Ktot + k0 + c8 * 8);
    }
}
__device__ inline void store_f16_tile(half_t* __restrict__ lds, int t, const v8h r[4]) {
#pragma unroll
    for (int q = 0; q < 4; ++q) {
        int c = t + 256 * q;
        int row = c >> 3, c8 = c & 7;
        *(v8h*)&lds[row * LDS_STR + c8 * 8] = r[q];
    }
}

// ---------------------------------------------------------------------------
// scale = 1 / (IN_F * sum(weights))
// ---------------------------------------------------------------------------
__global__ __launch_bounds__(256) void sumw_kernel(const float* __restrict__ w,
                                                   float* __restrict__ out) {
    __shared__ float s[256];
    s[threadIdx.x] = w[threadIdx.x] + w[threadIdx.x + 256];
    __syncthreads();
    for (int off = 128; off > 0; off >>= 1) {
        if ((int)threadIdx.x < off) s[threadIdx.x] += s[threadIdx.x + off];
        __syncthreads();
    }
    if (threadIdx.x == 0) out[0] = 1.0f / ((float)IN_F * s[0]);
}

// G1T[m][k] = N(x_in[k]; mu_m0, var_m0) * w[m]   (f16, [512][4096])
__global__ __launch_bounds__(256) void gen_g1(const float* __restrict__ mus,
                                              const float* __restrict__ log_vars,
                                              const float* __restrict__ w,
                                              half_t* __restrict__ G1T) {
    int idx = blockIdx.x * 256 + threadIdx.x;
    int m = idx >> 12, k = idx & 4095;
    float var = __expf(log_vars[2 * m + 0]);
    float mu  = mus[2 * m + 0];
    float d   = (float)k * (1.0f / 4095.0f) - mu;
    G1T[idx] = (half_t)(__expf(-d * d / (2.0f * var)) *
                        rsqrtf(6.283185307179586f * var) * w[m]);
}

// G2T[j][m] = N(x_out[j]; mu_m1, var_m1)   (f16, [4096][512]); global scale in epilogue
__global__ __launch_bounds__(256) void gen_g2(const float* __restrict__ mus,
                                              const float* __restrict__ log_vars,
                                              half_t* __restrict__ G2T) {
    int idx = blockIdx.x * 256 + threadIdx.x;
    int j = idx >> 9, m = idx & 511;
    float var = __expf(log_vars[2 * m + 1]);
    float mu  = mus[2 * m + 1];
    float d   = (float)j * (1.0f / 4095.0f) - mu;
    G2T[idx] = (half_t)(__expf(-d * d / (2.0f * var)) *
                        rsqrtf(6.283185307179586f * var));
}

// ---------------------------------------------------------------------------
// WMMA GEMM: Out[M x N] = A[M x K] * Bt^T   (Bt stored [N][K] f16)
// 256 threads = 8 waves (4x2); wave tile 32x64 = 2x4 WMMA tiles.
// Double-buffered LDS; B (and f16 A) staged by the Tensor Data Mover.
// ---------------------------------------------------------------------------
template <bool A_IS_F32, bool OUT_IS_F16, bool APPLY_SCALE, int Ktot, int Ntot>
__global__ __launch_bounds__(256) void gemm_wmma(const void* __restrict__ Ap,
                                                 const half_t* __restrict__ Bt,
                                                 void* __restrict__ Outp,
                                                 const float* __restrict__ scale_ptr) {
    __shared__ __attribute__((aligned(16))) half_t lds_a[2][128 * LDS_STR];
    __shared__ __attribute__((aligned(16))) half_t lds_b[2][128 * LDS_STR];

    const int t = threadIdx.x;
    const int wid = t >> 5, lane = t & 31;
    const int wm = wid & 3;          // wave row group: 32 rows
    const int wn = wid >> 2;         // wave col group: 64 cols
    const int half_id = lane >> 4, l16 = lane & 15;
    const int m0 = blockIdx.y * 128;
    const int n0 = blockIdx.x * 128;

    float outscale = 1.0f;
    if constexpr (APPLY_SCALE) outscale = scale_ptr[0];

    // ---- prologue: stage K-step 0 into buffer 0 ----
#if USE_TDM
    if (wid == 0) {
        tdm_tile_f16<KSTEP, 128, 4, 3>(Bt + (size_t)n0 * Ktot, lds_u32(&lds_b[0][0]), Ktot);
        if constexpr (!A_IS_F32)
            tdm_tile_f16<KSTEP, 128, 4, 3>((const half_t*)Ap + (size_t)m0 * Ktot,
                                           lds_u32(&lds_a[0][0]), Ktot);
    }
    if constexpr (A_IS_F32) {
        float4 r[8];
        loadA_f32<Ktot>((const float*)Ap, m0, 0, t, r);
        storeA_f32(&lds_a[0][0], t, r);
    }
    if (wid == 0) wait_tensor();
#else
    if constexpr (A_IS_F32) {
        float4 r[8];
        loadA_f32<Ktot>((const float*)Ap, m0, 0, t, r);
        storeA_f32(&lds_a[0][0], t, r);
    } else {
        v8h r[4];
        load_f16_tile<Ktot>((const half_t*)Ap, m0, 0, t, r);
        store_f16_tile(&lds_a[0][0], t, r);
    }
    {
        v8h r[4];
        load_f16_tile<Ktot>(Bt, n0, 0, t, r);
        store_f16_tile(&lds_b[0][0], t, r);
    }
#endif
    __syncthreads();

    v8f acc[2][4] = {};
    float4 aregs32[8];
    v8h    aregs16[4];
    v8h    bregs[4];

    for (int k0 = 0; k0 < Ktot; k0 += KSTEP) {
        const int cur = (k0 >> 6) & 1, nxt = cur ^ 1;
        const bool has_next = (k0 + KSTEP) < Ktot;

        // ---- kick off staging for the next K-step (overlaps the WMMAs) ----
        if (has_next) {
#if USE_TDM
            if (wid == 0) {
                tdm_tile_f16<KSTEP, 128, 4, 3>(Bt + (size_t)n0 * Ktot + k0 + KSTEP,
                                               lds_u32(&lds_b[nxt][0]), Ktot);
                if constexpr (!A_IS_F32)
                    tdm_tile_f16<KSTEP, 128, 4, 3>((const half_t*)Ap + (size_t)m0 * Ktot + k0 + KSTEP,
                                                   lds_u32(&lds_a[nxt][0]), Ktot);
            }
            if constexpr (A_IS_F32)
                loadA_f32<Ktot>((const float*)Ap, m0, k0 + KSTEP, t, aregs32);
#else
            if constexpr (A_IS_F32)
                loadA_f32<Ktot>((const float*)Ap, m0, k0 + KSTEP, t, aregs32);
            else
                load_f16_tile<Ktot>((const half_t*)Ap, m0, k0 + KSTEP, t, aregs16);
            load_f16_tile<Ktot>(Bt, n0, k0 + KSTEP, t, bregs);
#endif
        }

        // ---- 16 WMMAs from the current buffers ----
#pragma unroll
        for (int ks = 0; ks < KSTEP; ks += 32) {
            Frag16 a[2], b[4];
#pragma unroll
            for (int rt = 0; rt < 2; ++rt) {
                int row = wm * 32 + rt * 16 + l16;
                // lanes 0-15: K {0..7,16..23}; lanes 16-31: K {8..15,24..31}
                a[rt].h[0] = *(const v8h*)&lds_a[cur][row * LDS_STR + ks + half_id * 8];
                a[rt].h[1] = *(const v8h*)&lds_a[cur][row * LDS_STR + ks + 16 + half_id * 8];
            }
#pragma unroll
            for (int nt = 0; nt < 4; ++nt) {
                int n = wn * 64 + nt * 16 + l16;
                // lane holds column N=l16; lanes 0-15: K {0..7,16..23}; lanes 16-31: K {8..15,24..31}
                b[nt].h[0] = *(const v8h*)&lds_b[cur][n * LDS_STR + ks + half_id * 8];
                b[nt].h[1] = *(const v8h*)&lds_b[cur][n * LDS_STR + ks + 16 + half_id * 8];
            }
#pragma unroll
            for (int rt = 0; rt < 2; ++rt)
#pragma unroll
                for (int nt = 0; nt < 4; ++nt)
                    acc[rt][nt] = __builtin_amdgcn_wmma_f32_16x16x32_f16(
                        false, a[rt].v, false, b[nt].v, (short)0, acc[rt][nt],
                        false, false);
        }

        // ---- drain staging into the next buffer ----
        if (has_next) {
#if USE_TDM
            if constexpr (A_IS_F32) storeA_f32(&lds_a[nxt][0], t, aregs32);
            if (wid == 0) wait_tensor();
#else
            if constexpr (A_IS_F32) storeA_f32(&lds_a[nxt][0], t, aregs32);
            else                    store_f16_tile(&lds_a[nxt][0], t, aregs16);
            store_f16_tile(&lds_b[nxt][0], t, bregs);
#endif
        }
        __syncthreads();
    }

    // ---- epilogue: D element g -> row M = g + 8*half_id, col N = l16 ----
#pragma unroll
    for (int rt = 0; rt < 2; ++rt) {
#pragma unroll
        for (int nt = 0; nt < 4; ++nt) {
            int col = n0 + wn * 64 + nt * 16 + l16;
#pragma unroll
            for (int g = 0; g < 8; ++g) {
                int row = m0 + wm * 32 + rt * 16 + g + 8 * half_id;
                float o = acc[rt][nt][g] * outscale;
                if constexpr (OUT_IS_F16)
                    ((half_t*)Outp)[(size_t)row * Ntot + col] = (half_t)o;
                else
                    ((float*)Outp)[(size_t)row * Ntot + col] = o;
            }
        }
    }
}

// ---------------------------------------------------------------------------
extern "C" void kernel_launch(void* const* d_in, const int* in_sizes, int n_in,
                              void* d_out, int out_size, void* d_ws, size_t ws_size,
                              hipStream_t stream) {
    const float* x        = (const float*)d_in[0];   // [4096, 4096]
    const float* mus      = (const float*)d_in[1];   // [512, 2]
    const float* log_vars = (const float*)d_in[2];   // [512, 2]
    const float* weights  = (const float*)d_in[3];   // [512]
    float* out            = (float*)d_out;           // [4096, 4096]

    char* ws = (char*)d_ws;
    float*  scale = (float*)ws;                       // 1 float
    half_t* G1T   = (half_t*)(ws + 256);              // [512][4096] f16
    half_t* G2T   = G1T + (size_t)NCOMP * IN_F;       // [4096][512] f16
    half_t* C16   = G2T + (size_t)OUT_F * NCOMP;      // [4096][512] f16

    sumw_kernel<<<1, 256, 0, stream>>>(weights, scale);
    gen_g1<<<(NCOMP * IN_F) / 256, 256, 0, stream>>>(mus, log_vars, weights, G1T);
    gen_g2<<<(OUT_F * NCOMP) / 256, 256, 0, stream>>>(mus, log_vars, G2T);

    // C16 = (x @ g_in) * w     [4096 x 512]
    {
        dim3 grid(NCOMP / 128, BATCH / 128);
        gemm_wmma<true, true, false, IN_F, NCOMP>
            <<<grid, 256, 0, stream>>>(x, G1T, C16, scale);
    }
    // out = (C16 @ g_out) / (IN * sum_w)   [4096 x 4096]
    {
        dim3 grid(OUT_F / 128, BATCH / 128);
        gemm_wmma<false, false, true, NCOMP, OUT_F>
            <<<grid, 256, 0, stream>>>(C16, G2T, out, scale);
    }
}
